// GrassmannProjector_69715909149466
// MI455X (gfx1250) — matrix operand order repaired and checked by reading, not compiled, not run on hardware
//
#include <hip/hip_runtime.h>

// ---------------------------------------------------------------------------
// GrassmannProjector on MI455X (gfx1250):
//   out[b,k,:] = sigma_k * Vh[b,k,:]  ==  sqrt(lambda_k) * v_k  where
//   (lambda_k, v_k) are top-8 eigenpairs of G = x^T x  (256x256 per batch).
// Stage 1: G = x^T x with v_wmma_f32_16x16x32_bf16 (f32 accumulate),
//          double-buffered LDS slab, packed bf16x2 ds_store_b32 staging.
// Stage 2: subspace iteration + MGS on 256x8 block, per-batch workgroup.
// ---------------------------------------------------------------------------

typedef __attribute__((ext_vector_type(16))) __bf16 v16bf;
typedef __attribute__((ext_vector_type(8)))  __bf16 bf16x8;
typedef __attribute__((ext_vector_type(2)))  __bf16 bf16x2;
typedef __attribute__((ext_vector_type(8)))  float  v8f;

#define BATCH 128
#define KDIM  1024
#define NDIM  256
#define TOPK  8
#define SSTR  40      // padded LDS column stride in bf16 elems (80 B, 16B-aligned)
#define NITER 24

// Stage a 32(k) x 256(c) slab of x into LDS, transposed to slab[c*SSTR+k],
// converting f32 -> bf16 and packing k-pairs into 4-byte ds_store_b32.
__device__ __forceinline__ void stage_slab(
    const float* __restrict__ xb, int k0, __bf16* __restrict__ dst,
    int cg, int rp)
{
#pragma unroll
  for (int i = 0; i < 4; ++i) {
    const int r0 = 2 * (rp + 4 * i);       // even k within slab
    const float4 a = *(const float4*)(xb + (size_t)(k0 + r0) * NDIM + cg);
    const float4 b = *(const float4*)(xb + (size_t)(k0 + r0 + 1) * NDIM + cg);
    *(bf16x2*)(dst + (cg + 0) * SSTR + r0) = bf16x2{(__bf16)a.x, (__bf16)b.x};
    *(bf16x2*)(dst + (cg + 1) * SSTR + r0) = bf16x2{(__bf16)a.y, (__bf16)b.y};
    *(bf16x2*)(dst + (cg + 2) * SSTR + r0) = bf16x2{(__bf16)a.z, (__bf16)b.z};
    *(bf16x2*)(dst + (cg + 3) * SSTR + r0) = bf16x2{(__bf16)a.w, (__bf16)b.w};
  }
}

// ========================= Stage 1: Gram via WMMA ==========================
__global__ __launch_bounds__(256) void gram_wmma_kernel(
    const float* __restrict__ x, float* __restrict__ G)
{
  const int b    = blockIdx.x;          // batch
  const int i0   = blockIdx.y * 16;     // G row tile origin
  const int t    = threadIdx.x;
  const int lane = t & 31;
  const int wave = t >> 5;              // 8 waves
  const int m    = lane & 15;           // M (A) / N (B) index within tile
  const int g    = lane >> 4;           // half-wave group (K split)

  const float* xb = x + (size_t)b * KDIM * NDIM;
  float*       Gb = G + (size_t)b * NDIM * NDIM;

  // Double-buffered transposed bf16 slab: 2 x 20 KB (WGP has 320 KB LDS)
  __shared__ __align__(16) __bf16 slab[2][NDIM * SSTR];

  v8f acc0 = {};   // tile (i0, wave*16)
  v8f acc1 = {};   // tile (i0, (wave+8)*16)

  const int cg = (t & 63) * 4;   // column group (4 floats, coalesced)
  const int rp = t >> 6;         // row-pair base (0..3)

  // Prologue: fill buffer 0
  stage_slab(xb, 0, slab[0], cg, rp);

  int p = 0;
  for (int k0 = 0; k0 < KDIM; k0 += 32, p ^= 1) {
    __syncthreads();   // slab[p] ready; slab[p^1] free to overwrite

    // Prefetch next slab into the other buffer (overlaps with WMMAs below)
    if (k0 + 32 < KDIM) stage_slab(xb, k0 + 32, slab[p ^ 1], cg, rp);

    const __bf16* s = slab[p];

    // A fragment (16x32 bf16): lane row M=m, elems e -> K = e + 16g
    // (contiguous K order, matching B)
    const __bf16* colA = s + (i0 + m) * SSTR;
    bf16x8 alo = *(const bf16x8*)(colA + 16 * g);
    bf16x8 ahi = *(const bf16x8*)(colA + 16 * g + 8);
    v16bf afrag = __builtin_shufflevector(alo, ahi,
        0,1,2,3,4,5,6,7,8,9,10,11,12,13,14,15);

    // B fragments (32x16 bf16): lane col N=m, elems e -> K = e + 16g
    {
      const __bf16* colB = s + (wave * 16 + m) * SSTR;
      bf16x8 b0 = *(const bf16x8*)(colB + 16 * g);
      bf16x8 b1 = *(const bf16x8*)(colB + 16 * g + 8);
      v16bf bfrag = __builtin_shufflevector(b0, b1,
          0,1,2,3,4,5,6,7,8,9,10,11,12,13,14,15);
      acc0 = __builtin_amdgcn_wmma_f32_16x16x32_bf16(
          false, afrag, false, bfrag, (short)0, acc0, false, false);
    }
    {
      const __bf16* colB = s + ((wave + 8) * 16 + m) * SSTR;
      bf16x8 b0 = *(const bf16x8*)(colB + 16 * g);
      bf16x8 b1 = *(const bf16x8*)(colB + 16 * g + 8);
      v16bf bfrag = __builtin_shufflevector(b0, b1,
          0,1,2,3,4,5,6,7,8,9,10,11,12,13,14,15);
      acc1 = __builtin_amdgcn_wmma_f32_16x16x32_bf16(
          false, afrag, false, bfrag, (short)0, acc1, false, false);
    }
  }

  // C/D layout: VGPR r -> M = r + 8g, N = m
#pragma unroll
  for (int r = 0; r < 8; ++r) {
    const int row = i0 + r + 8 * g;
    Gb[(size_t)row * NDIM + wave * 16 + m]       = acc0[r];
    Gb[(size_t)row * NDIM + (wave + 8) * 16 + m] = acc1[r];
  }
}

// ==================== Stage 2: top-8 eigenpairs of G =======================
__device__ __forceinline__ float wave_reduce_add(float v) {
#pragma unroll
  for (int off = 16; off > 0; off >>= 1) v += __shfl_xor(v, off, 32);
  return v;
}

__global__ __launch_bounds__(256) void topk_eig_kernel(
    const float* __restrict__ G, float* __restrict__ out)
{
  const int b    = blockIdx.x;
  const int r    = threadIdx.x;      // row of Q owned by this thread
  const int lane = r & 31;
  const int wave = r >> 5;

  const float* Gb = G + (size_t)b * NDIM * NDIM;

  __shared__ float Qs[NDIM][TOPK + 1];   // +1 pad vs bank conflicts
  __shared__ float red[8 * TOPK];        // cross-wave reduction scratch
  __shared__ float lamS[TOPK];
  __shared__ float sgnS[TOPK];
  __shared__ int   permS[TOPK];

  // Deterministic pseudo-random init of Q
#pragma unroll
  for (int j = 0; j < TOPK; ++j) {
    unsigned h = (unsigned)r * 2654435761u + (unsigned)j * 40503u + 0x9E3779B9u;
    h ^= h >> 13; h *= 0x85EBCA6Bu; h ^= h >> 16;
    Qs[r][j] = ((float)(h & 0xFFFFu) * (1.0f / 65536.0f)) - 0.5f;
  }
  __syncthreads();

  float y[TOPK];
  for (int it = 0; it <= NITER; ++it) {
    // y(row r) = (G * Q)[r, :]
#pragma unroll
    for (int j = 0; j < TOPK; ++j) y[j] = 0.0f;
    const float4* gr = (const float4*)(Gb + (size_t)r * NDIM);
    for (int c4 = 0; c4 < NDIM / 4; ++c4) {
      const float4 gv = gr[c4];
      const int c = c4 * 4;
#pragma unroll
      for (int j = 0; j < TOPK; ++j) {
        y[j] += gv.x * Qs[c + 0][j] + gv.y * Qs[c + 1][j]
              + gv.z * Qs[c + 2][j] + gv.w * Qs[c + 3][j];
      }
    }
    if (it == NITER) break;   // keep y = G*Q for Rayleigh quotients

    // Modified Gram-Schmidt on the 8 columns of y
    for (int j = 0; j < TOPK; ++j) {
      // ||y_j||
      float p = wave_reduce_add(y[j] * y[j]);
      __syncthreads();
      if (lane == 0) red[wave] = p;
      __syncthreads();
      float nn = 0.0f;
#pragma unroll
      for (int w = 0; w < 8; ++w) nn += red[w];
      const float inv = rsqrtf(nn + 1e-30f);
      y[j] *= inv;

      // batched dots <q_j, y_l> for l > j
      float pj[TOPK];
#pragma unroll
      for (int l = 0; l < TOPK; ++l) pj[l] = 0.0f;
      for (int l = j + 1; l < TOPK; ++l) pj[l] = wave_reduce_add(y[j] * y[l]);
      __syncthreads();
      if (lane == 0) {
#pragma unroll
        for (int l = 0; l < TOPK; ++l) red[wave * 8 + l] = pj[l];
      }
      __syncthreads();
      for (int l = j + 1; l < TOPK; ++l) {
        float d = 0.0f;
#pragma unroll
        for (int w = 0; w < 8; ++w) d += red[w * 8 + l];
        y[l] -= d * y[j];
      }
    }
    __syncthreads();
#pragma unroll
    for (int j = 0; j < TOPK; ++j) Qs[r][j] = y[j];
    __syncthreads();
  }

  // Rayleigh quotients lambda_j = q_j^T G q_j ; sign convention from sum(q_j)
  for (int j = 0; j < TOPK; ++j) {
    float p  = wave_reduce_add(Qs[r][j] * y[j]);
    float sp = wave_reduce_add(Qs[r][j]);
    __syncthreads();
    if (lane == 0) { red[wave] = p; red[8 + wave] = sp; }
    __syncthreads();
    float lam = 0.0f, sg = 0.0f;
#pragma unroll
    for (int w = 0; w < 8; ++w) { lam += red[w]; sg += red[8 + w]; }
    if (r == 0) { lamS[j] = lam; sgnS[j] = (sg >= 0.0f) ? 1.0f : -1.0f; }
    __syncthreads();
  }

  // Sort the 8 eigenvalues descending (thread 0, tiny bubble sort)
  if (r == 0) {
    for (int j = 0; j < TOPK; ++j) permS[j] = j;
    for (int a = 0; a < TOPK - 1; ++a)
      for (int c = 0; c < TOPK - 1 - a; ++c)
        if (lamS[permS[c]] < lamS[permS[c + 1]]) {
          int tmp = permS[c]; permS[c] = permS[c + 1]; permS[c + 1] = tmp;
        }
  }
  __syncthreads();

  // out[b, jj, r] = sqrt(lambda) * sign * v[r]
#pragma unroll
  for (int jj = 0; jj < TOPK; ++jj) {
    const int j = permS[jj];
    const float s = sqrtf(fmaxf(lamS[j], 0.0f)) * sgnS[j];
    out[(size_t)b * TOPK * NDIM + (size_t)jj * NDIM + r] = s * Qs[r][j];
  }
}

// =============================== launcher ==================================
extern "C" void kernel_launch(void* const* d_in, const int* in_sizes, int n_in,
                              void* d_out, int out_size, void* d_ws, size_t ws_size,
                              hipStream_t stream) {
  (void)in_sizes; (void)n_in; (void)out_size; (void)ws_size;
  const float* x = (const float*)d_in[0];
  float* out = (float*)d_out;
  float* G   = (float*)d_ws;          // 128 * 256 * 256 * 4 = 32 MB

  dim3 grid1(BATCH, NDIM / 16);
  gram_wmma_kernel<<<grid1, 256, 0, stream>>>(x, G);
  topk_eig_kernel<<<BATCH, 256, 0, stream>>>(G, out);
}
